// GcnInfomax_12232066859169
// MI455X (gfx1250) — matrix operand order repaired and checked by reading, not compiled, not run on hardware
//
#include <hip/hip_runtime.h>
#include <hip/hip_bf16.h>

// ---------------------------------------------------------------------------
// GcnInfomax fused pipeline for MI455X (gfx1250, wave32, WMMA f16->f32)
//   Stage 0: f32 -> f16 conversions (inputs) + weight transpose/convert
//   Stage 1: 8x FF GEMMs [8192x256]x[256x256] via v_wmma_f32_16x16x32_f16
//   Stage 2: fused res = l_enc @ g_enc^T + masked JSD reduction (no 268MB
//            res materialization; adj streamed with non-temporal loads)
//   Stage 3: deterministic final reduction -> scalar
// ---------------------------------------------------------------------------

typedef _Float16 f16;
typedef __attribute__((ext_vector_type(16))) _Float16 v16h;
typedef __attribute__((ext_vector_type(8)))  _Float16 v8h;
typedef __attribute__((ext_vector_type(8)))  float    v8f;

#define NROWS 8192
#define KDIM  256
#define LOG2F_C 0.6931471805599453f

// ---- fragment loaders matching CDNA5 ISA 16-bit WMMA VGPR layouts ----------
// A (16x32, MxK): lane L: m = L%16, half = L/16.
//   a[0..7]  = K (kk + half*8     + j)
//   a[8..15] = K (kk + 16+half*8 + j-8)
__device__ __forceinline__ v16h load_a_frag(const f16* __restrict__ A,
                                            int row0, int kk, int lane) {
  int m = lane & 15, half = lane >> 4;
  const f16* p = A + (size_t)(row0 + m) * KDIM + kk + half * 8;
  v8h lo = *(const v8h*)p;
  v8h hi = *(const v8h*)(p + 16);
  v16h a;
#pragma unroll
  for (int j = 0; j < 8; ++j) { a[j] = lo[j]; a[j + 8] = hi[j]; }
  return a;
}

// B (32x16, KxN): lane L: n = L%16, half = L/16.
//   b[0..7]  = K (kk + half*8     + j)
//   b[8..15] = K (kk + 16+half*8 + j-8)
// Source matrix is row-major [n][k] (i.e. output-column-major), so this is
// two 16-byte loads per lane.
__device__ __forceinline__ v16h load_b_frag(const f16* __restrict__ B,
                                            int n0, int kk, int lane) {
  int n = lane & 15, half = lane >> 4;
  const f16* p = B + (size_t)(n0 + n) * KDIM + kk + half * 8;
  v8h lo = *(const v8h*)p;
  v8h hi = *(const v8h*)(p + 16);
  v16h b;
#pragma unroll
  for (int j = 0; j < 8; ++j) { b[j] = lo[j]; b[j + 8] = hi[j]; }
  return b;
}

__device__ __forceinline__ float softplusf(float x) {
  // stable softplus: max(x,0) + log1p(exp(-|x|))
  return fmaxf(x, 0.f) + log1pf(__expf(-fabsf(x)));
}

// ---- stage 0 ----------------------------------------------------------------
__global__ void cvt_f16_kernel(const float* __restrict__ in, f16* __restrict__ out, int n) {
  int i = blockIdx.x * blockDim.x + threadIdx.x;
  if (i < n) out[i] = (f16)in[i];
}

// Wt[n][k] = (f16) W[k][n]   (W is fan_in x fan_out = [k][n], 256x256)
__global__ void wtrans_kernel(const float* __restrict__ W, f16* __restrict__ Wt) {
  int k = blockIdx.x;
  int n = threadIdx.x;
  Wt[(size_t)n * KDIM + k] = (f16)W[(size_t)k * KDIM + n];
}

// ---- stage 1: Out[16-row tile] = maybe_acc + maybe_relu(A @ W + b) ----------
// grid.x = 512 row tiles, block = 256 threads (8 waves), each wave: 2 col tiles
__global__ void ff_gemm_kernel(const f16* __restrict__ A, const f16* __restrict__ Wt,
                               const float* __restrict__ bias, f16* __restrict__ Out,
                               int do_relu, int do_acc) {
  int lane = threadIdx.x & 31;
  int wave = threadIdx.x >> 5;
  int row0 = blockIdx.x * 16;
  int n0 = wave * 32;

  v8f c0 = {}; v8f c1 = {};
#pragma unroll
  for (int kk = 0; kk < KDIM; kk += 32) {
    v16h a  = load_a_frag(A, row0, kk, lane);
    v16h b0 = load_b_frag(Wt, n0,      kk, lane);
    v16h b1 = load_b_frag(Wt, n0 + 16, kk, lane);
    c0 = __builtin_amdgcn_wmma_f32_16x16x32_f16(false, a, false, b0, (short)0, c0, false, false);
    c1 = __builtin_amdgcn_wmma_f32_16x16x32_f16(false, a, false, b1, (short)0, c1, false, false);
  }

  int nl = lane & 15, half = lane >> 4;
  float bv0 = bias[n0 + nl];
  float bv1 = bias[n0 + 16 + nl];
#pragma unroll
  for (int r = 0; r < 8; ++r) {
    int row = row0 + r + half * 8;
    float v0 = c0[r] + bv0;
    float v1 = c1[r] + bv1;
    if (do_relu) { v0 = fmaxf(v0, 0.f); v1 = fmaxf(v1, 0.f); }
    f16* o0 = Out + (size_t)row * KDIM + n0 + nl;
    f16* o1 = o0 + 16;
    if (do_acc) { v0 += (float)(*o0); v1 += (float)(*o1); }
    *o0 = (f16)v0;
    *o1 = (f16)v1;
  }
}

// ---- stage 2: fused pairwise GEMM + masked JSD partial sums -----------------
// grid = (512, 16), block = 256 (8 waves). Wave: 1 row tile x 4 col tiles.
__global__ void pairwise_jsd_kernel(const f16* __restrict__ L, const f16* __restrict__ G,
                                    const float* __restrict__ adj,
                                    float* __restrict__ pos_part,
                                    float* __restrict__ neg_part) {
  int lane = threadIdx.x & 31;
  int wave = threadIdx.x >> 5;
  int row0 = blockIdx.x * 16;
  int j0 = (blockIdx.y * 32 + wave * 4) * 16;

  v8f c[4] = {};
#pragma unroll
  for (int kk = 0; kk < KDIM; kk += 32) {
    v16h a = load_a_frag(L, row0, kk, lane);
#pragma unroll
    for (int t = 0; t < 4; ++t) {
      v16h b = load_b_frag(G, j0 + t * 16, kk, lane);
      c[t] = __builtin_amdgcn_wmma_f32_16x16x32_f16(false, a, false, b, (short)0, c[t], false, false);
    }
  }

  int nl = lane & 15, half = lane >> 4;
  float pos = 0.f, neg = 0.f;
#pragma unroll
  for (int t = 0; t < 4; ++t) {
    int col = j0 + t * 16 + nl;
#pragma unroll
    for (int r = 0; r < 8; ++r) {
      int row = row0 + r + half * 8;
      float res = c[t][r];
      // non-temporal: don't let the 268MB adj stream evict L2-resident encs
      float aij = __builtin_nontemporal_load(&adj[(size_t)row * NROWS + col]);
      float p = res * aij;        // res * pos_mask
      float q = res - p;          // res * (1 - pos_mask)
      pos += LOG2F_C - softplusf(-p);
      neg += softplusf(-q) + q - LOG2F_C;
    }
  }

  __shared__ float sp[256], sn[256];
  sp[threadIdx.x] = pos;
  sn[threadIdx.x] = neg;
  __syncthreads();
#pragma unroll
  for (int s = 128; s > 0; s >>= 1) {
    if (threadIdx.x < (unsigned)s) {
      sp[threadIdx.x] += sp[threadIdx.x + s];
      sn[threadIdx.x] += sn[threadIdx.x + s];
    }
    __syncthreads();
  }
  if (threadIdx.x == 0) {
    int bid = blockIdx.y * gridDim.x + blockIdx.x;   // 0..8191
    pos_part[bid] = sp[0];
    neg_part[bid] = sn[0];
  }
}

// ---- stage 3: deterministic final reduction ---------------------------------
__global__ void finalize_kernel(const float* __restrict__ pos_part,
                                const float* __restrict__ neg_part,
                                float* __restrict__ out) {
  __shared__ float sp[256], sn[256];
  float p = 0.f, n = 0.f;
  for (int i = threadIdx.x; i < 8192; i += 256) { p += pos_part[i]; n += neg_part[i]; }
  sp[threadIdx.x] = p;
  sn[threadIdx.x] = n;
  __syncthreads();
#pragma unroll
  for (int s = 128; s > 0; s >>= 1) {
    if (threadIdx.x < (unsigned)s) {
      sp[threadIdx.x] += sp[threadIdx.x + s];
      sn[threadIdx.x] += sn[threadIdx.x + s];
    }
    __syncthreads();
  }
  if (threadIdx.x == 0) {
    double Nd = (double)NROWS;
    double e_pos = (double)sp[0] / Nd;
    double e_neg = (double)sn[0] / (Nd * (Nd - 1.0));
    out[0] = (float)(e_neg - e_pos);
  }
}

// ---------------------------------------------------------------------------
extern "C" void kernel_launch(void* const* d_in, const int* in_sizes, int n_in,
                              void* d_out, int out_size, void* d_ws, size_t ws_size,
                              hipStream_t stream) {
  (void)in_sizes; (void)n_in; (void)out_size; (void)ws_size;

  const float* emb  = (const float*)d_in[0];
  const float* feat = (const float*)d_in[1];
  const float* adj  = (const float*)d_in[2];
  // d_in[3] = num_drugs (int, ==8192, hardcoded)
  const float* g_w1 = (const float*)d_in[4];  const float* g_b1 = (const float*)d_in[5];
  const float* g_w2 = (const float*)d_in[6];  const float* g_b2 = (const float*)d_in[7];
  const float* g_w3 = (const float*)d_in[8];  const float* g_b3 = (const float*)d_in[9];
  const float* g_ws = (const float*)d_in[10]; const float* g_bs = (const float*)d_in[11];
  const float* l_w1 = (const float*)d_in[12]; const float* l_b1 = (const float*)d_in[13];
  const float* l_w2 = (const float*)d_in[14]; const float* l_b2 = (const float*)d_in[15];
  const float* l_w3 = (const float*)d_in[16]; const float* l_b3 = (const float*)d_in[17];
  const float* l_ws = (const float*)d_in[18]; const float* l_bs = (const float*)d_in[19];

  // ---- workspace layout ----
  const size_t SZ_MAT16 = (size_t)NROWS * KDIM * sizeof(f16);   // 4 MB
  const size_t SZ_W16   = (size_t)KDIM * KDIM * sizeof(f16);    // 128 KB
  char* ws = (char*)d_ws;
  f16* emb16  = (f16*)(ws + 256);
  f16* feat16 = (f16*)(ws + 256 + 1 * SZ_MAT16);
  f16* hA     = (f16*)(ws + 256 + 2 * SZ_MAT16);
  f16* hB     = (f16*)(ws + 256 + 3 * SZ_MAT16);
  f16* genc   = (f16*)(ws + 256 + 4 * SZ_MAT16);
  f16* lenc   = (f16*)(ws + 256 + 5 * SZ_MAT16);
  char* wbase = ws + 256 + 6 * SZ_MAT16;
  f16* g_w1t = (f16*)(wbase + 0 * SZ_W16);
  f16* g_w2t = (f16*)(wbase + 1 * SZ_W16);
  f16* g_w3t = (f16*)(wbase + 2 * SZ_W16);
  f16* g_wst = (f16*)(wbase + 3 * SZ_W16);
  f16* l_w1t = (f16*)(wbase + 4 * SZ_W16);
  f16* l_w2t = (f16*)(wbase + 5 * SZ_W16);
  f16* l_w3t = (f16*)(wbase + 6 * SZ_W16);
  f16* l_wst = (f16*)(wbase + 7 * SZ_W16);
  float* pos_part = (float*)(wbase + 8 * SZ_W16);
  float* neg_part = pos_part + 8192;

  const int NMAT = NROWS * KDIM;

  // stage 0: conversions
  cvt_f16_kernel<<<NMAT / 256, 256, 0, stream>>>(emb,  emb16,  NMAT);
  cvt_f16_kernel<<<NMAT / 256, 256, 0, stream>>>(feat, feat16, NMAT);
  wtrans_kernel<<<KDIM, KDIM, 0, stream>>>(g_w1, g_w1t);
  wtrans_kernel<<<KDIM, KDIM, 0, stream>>>(g_w2, g_w2t);
  wtrans_kernel<<<KDIM, KDIM, 0, stream>>>(g_w3, g_w3t);
  wtrans_kernel<<<KDIM, KDIM, 0, stream>>>(g_ws, g_wst);
  wtrans_kernel<<<KDIM, KDIM, 0, stream>>>(l_w1, l_w1t);
  wtrans_kernel<<<KDIM, KDIM, 0, stream>>>(l_w2, l_w2t);
  wtrans_kernel<<<KDIM, KDIM, 0, stream>>>(l_w3, l_w3t);
  wtrans_kernel<<<KDIM, KDIM, 0, stream>>>(l_ws, l_wst);

  const int RT = NROWS / 16;  // 512 row tiles

  // stage 1: global encoder  genc = relu3(...) + emb @ Ws + bs
  ff_gemm_kernel<<<RT, 256, 0, stream>>>(emb16, g_wst, g_bs, genc, 0, 0); // shortcut
  ff_gemm_kernel<<<RT, 256, 0, stream>>>(emb16, g_w1t, g_b1, hA,   1, 0);
  ff_gemm_kernel<<<RT, 256, 0, stream>>>(hA,    g_w2t, g_b2, hB,   1, 0);
  ff_gemm_kernel<<<RT, 256, 0, stream>>>(hB,    g_w3t, g_b3, genc, 1, 1); // += relu(h2 W3 + b3)

  // stage 1: local encoder
  ff_gemm_kernel<<<RT, 256, 0, stream>>>(feat16, l_wst, l_bs, lenc, 0, 0);
  ff_gemm_kernel<<<RT, 256, 0, stream>>>(feat16, l_w1t, l_b1, hA,   1, 0);
  ff_gemm_kernel<<<RT, 256, 0, stream>>>(hA,     l_w2t, l_b2, hB,   1, 0);
  ff_gemm_kernel<<<RT, 256, 0, stream>>>(hB,     l_w3t, l_b3, lenc, 1, 1);

  // stage 2: fused pairwise scores + masked JSD partials
  dim3 grid2(RT, 16);
  pairwise_jsd_kernel<<<grid2, 256, 0, stream>>>(lenc, genc, adj, pos_part, neg_part);

  // stage 3: deterministic scalar
  finalize_kernel<<<1, 256, 0, stream>>>(pos_part, neg_part, (float*)d_out);
}
